// MLP_4337916970028
// MI455X (gfx1250) — compile-verified, matrix operand order checked by reading, not run on hardware
//
#include <hip/hip_runtime.h>
#include <hip/hip_bf16.h>
#include <math.h>

#define TOK   16384
#define EMB   1024
#define HID   4096
#define NN    8192
#define INV_SQRT_N 0.011048543456039805f   /* 1/sqrt(8192) */

typedef __attribute__((ext_vector_type(16))) __bf16 v16bf;
typedef __attribute__((ext_vector_type(8)))  float  v8f;
typedef __attribute__((ext_vector_type(4)))  int    v4i;
using bf16 = __hip_bfloat16;

// ---- CDNA5 async global->LDS path (guarded; falls back to sync copy) ------
#if defined(__HIP_DEVICE_COMPILE__) && defined(__has_builtin)
#if __has_builtin(__builtin_amdgcn_global_load_async_to_lds_b128)
#define USE_ASYNC_LDS 1
#endif
#endif

#ifdef USE_ASYNC_LDS
#define AS1 __attribute__((address_space(1)))
#define AS3 __attribute__((address_space(3)))
__device__ __forceinline__ void async_b128(const void* g, void* l) {
  __builtin_amdgcn_global_load_async_to_lds_b128((AS1 v4i*)(g), (AS3 v4i*)(l), 0, 0);
}
#if __has_builtin(__builtin_amdgcn_s_wait_asynccnt)
#define WAIT_ASYNC() __builtin_amdgcn_s_wait_asynccnt(0)
#else
#define WAIT_ASYNC() asm volatile("s_wait_asynccnt 0x0" ::: "memory")
#endif
#else
#define WAIT_ASYNC() ((void)0)
#endif

// ---------------------------------------------------------------------------
// Tiled transpose: w_down [NN, HID] -> w2T [HID, NN]
// ---------------------------------------------------------------------------
__global__ __launch_bounds__(256) void k_transpose(const float* __restrict__ in,
                                                   float* __restrict__ out) {
  __shared__ float tile[32][33];
  const int bx = blockIdx.x * 32;   // HID coord
  const int by = blockIdx.y * 32;   // NN coord
  const int tx = threadIdx.x;       // 0..31
  const int ty = threadIdx.y;       // 0..7
#pragma unroll
  for (int k = 0; k < 32; k += 8)
    tile[ty + k][tx] = in[(size_t)(by + ty + k) * HID + bx + tx];
  __syncthreads();
#pragma unroll
  for (int k = 0; k < 32; k += 8)
    out[(size_t)(bx + ty + k) * NN + by + tx] = tile[tx][ty + k];
}

// ---------------------------------------------------------------------------
// Fused: row-wise FWHT (length 8192) + sparse scatter into 1024 bins.
// MODE 0: write up^T  [HID, EMB] (row-contiguous)   -> B operand of GEMM1
// MODE 1: write down  [EMB, HID] (scattered column) -> B operand of GEMM2
// ---------------------------------------------------------------------------
template <int MODE>
__global__ __launch_bounds__(256) void k_fwht_scatter(const float* __restrict__ W,     // [HID, NN]
                                                      const float* __restrict__ sign,  // [NN]
                                                      const int*   __restrict__ prows, // [NN]
                                                      const float* __restrict__ pvals, // [NN]
                                                      bf16* __restrict__ outp) {
  __shared__ float data[NN];
  __shared__ float bins[EMB];
  const int d   = blockIdx.x;
  const int tid = threadIdx.x;

  const float4* W4 = reinterpret_cast<const float4*>(W + (size_t)d * NN);
  const float4* S4 = reinterpret_cast<const float4*>(sign);
  for (int i = tid; i < NN / 4; i += 256) {
    float4 w = W4[i], s = S4[i];
    data[4 * i + 0] = w.x * s.x;
    data[4 * i + 1] = w.y * s.y;
    data[4 * i + 2] = w.z * s.z;
    data[4 * i + 3] = w.w * s.w;
  }
  for (int i = tid; i < EMB; i += 256) bins[i] = 0.0f;
  __syncthreads();

  // 13 butterfly stages, natural (Sylvester) ordering
  for (int h = 1; h < NN; h <<= 1) {
    for (int p = tid; p < NN / 2; p += 256) {
      int i = ((p & ~(h - 1)) << 1) | (p & (h - 1));
      float a = data[i], b = data[i + h];
      data[i] = a + b;
      data[i + h] = a - b;
    }
    __syncthreads();
  }

  // scatter: bins[rows[k]] += v[k] * scale * data[k]
  for (int k = tid; k < NN; k += 256) {
    int r = prows[k];
    atomicAdd(&bins[r], pvals[k] * (INV_SQRT_N * data[k]));
  }
  __syncthreads();

  if (MODE == 0) {
    for (int r = tid; r < EMB; r += 256)
      outp[(size_t)d * EMB + r] = __float2bfloat16(bins[r]);
  } else {
    for (int r = tid; r < EMB; r += 256)
      outp[(size_t)r * HID + d] = __float2bfloat16(bins[r]);
  }
}

// ---------------------------------------------------------------------------
// x f32 -> bf16 (vectorized)
// ---------------------------------------------------------------------------
__global__ __launch_bounds__(256) void k_f32_to_bf16(const float4* __restrict__ in,
                                                     uint2* __restrict__ out, int n4) {
  int i = blockIdx.x * blockDim.x + threadIdx.x;
  if (i < n4) {
    float4 v = in[i];
    union { bf16 h[4]; uint2 u; } o;
    o.h[0] = __float2bfloat16(v.x);
    o.h[1] = __float2bfloat16(v.y);
    o.h[2] = __float2bfloat16(v.z);
    o.h[3] = __float2bfloat16(v.w);
    out[i] = o.u;
  }
}

// ---------------------------------------------------------------------------
// bf16 WMMA GEMM:  C[M,N] = A[M,K] * B[N,K]^T   (both operands row-major [*,K])
// Block tile 128x256x32, 8 waves (2x4), wave tile 64x64 = 4x4 WMMA frags.
// Double-buffered LDS, async global->LDS fills, 1 barrier per K-step.
// GELU epilogue + bf16 store for GEMM1; f32 store for GEMM2.
// ---------------------------------------------------------------------------
template <bool GELU>
__global__ __launch_bounds__(256) void k_gemm(const bf16* __restrict__ A,
                                              const bf16* __restrict__ B,
                                              int M, int N, int K,
                                              float* __restrict__ Cf,
                                              bf16*  __restrict__ Cb) {
  constexpr int LDT = 40;  // LDS row pitch in bf16 (80B = 5 uint4, 16B-aligned)
  __shared__ __align__(16) bf16 As[2][128 * LDT];   // 2 x 10 KB
  __shared__ __align__(16) bf16 Bs[2][256 * LDT];   // 2 x 20 KB

  const int tid   = threadIdx.x;
  const int wave  = tid >> 5;
  const int lane  = tid & 31;
  const int half  = lane >> 4;   // K-subchunk selector per ISA layout
  const int l16   = lane & 15;   // M (A) / N (B) index within 16x16 tile
  const int waveM = (wave & 1) * 64;
  const int waveN = (wave >> 1) * 64;

  // global->LDS copy mapping (per fill: A thread moves 2 uint4, B thread moves 4)
  const size_t Kq = (size_t)K >> 3;  // row pitch in uint4
  const uint4* Ag = reinterpret_cast<const uint4*>(A) +
                    (size_t)(blockIdx.y * 128 + (tid >> 1)) * Kq + (tid & 1) * 2;
  const uint4* Bg = reinterpret_cast<const uint4*>(B) +
                    (size_t)(blockIdx.x * 256 + tid) * Kq;
  uint4* AsQ[2] = {reinterpret_cast<uint4*>(&As[0][0]) + (tid >> 1) * 5 + (tid & 1) * 2,
                   reinterpret_cast<uint4*>(&As[1][0]) + (tid >> 1) * 5 + (tid & 1) * 2};
  uint4* BsQ[2] = {reinterpret_cast<uint4*>(&Bs[0][0]) + tid * 5,
                   reinterpret_cast<uint4*>(&Bs[1][0]) + tid * 5};

  auto fill = [&](int buf, int kt) {
    const uint4* ag = Ag + (size_t)kt * 4;
    const uint4* bg = Bg + (size_t)kt * 4;
    uint4* aq = AsQ[buf];
    uint4* bq = BsQ[buf];
#ifdef USE_ASYNC_LDS
    async_b128(ag + 0, aq + 0);
    async_b128(ag + 1, aq + 1);
    async_b128(bg + 0, bq + 0);
    async_b128(bg + 1, bq + 1);
    async_b128(bg + 2, bq + 2);
    async_b128(bg + 3, bq + 3);
#else
    uint4 a0 = ag[0], a1 = ag[1];
    uint4 b0 = bg[0], b1 = bg[1], b2 = bg[2], b3 = bg[3];
    aq[0] = a0; aq[1] = a1;
    bq[0] = b0; bq[1] = b1; bq[2] = b2; bq[3] = b3;
#endif
  };

  v8f acc[4][4];
#pragma unroll
  for (int i = 0; i < 4; ++i)
#pragma unroll
    for (int j = 0; j < 4; ++j)
#pragma unroll
      for (int e = 0; e < 8; ++e) acc[i][j][e] = 0.0f;

  const int ksteps = K >> 5;
  fill(0, 0);

  for (int kt = 0; kt < ksteps; ++kt) {
    const int cur = kt & 1;
    WAIT_ASYNC();        // my async fills of buf[cur] have landed in LDS
    __syncthreads();     // everyone's fills landed; everyone done reading buf[1-cur]

    const bf16* Ab = As[cur];
    const bf16* Bb = Bs[cur];
    union Q { uint4 q[2]; v16bf v; };
    v16bf af[4], bfr[4];
#pragma unroll
    for (int i = 0; i < 4; ++i) {
      const int m = waveM + i * 16 + l16;
      Q u;
      u.q[0] = *reinterpret_cast<const uint4*>(Ab + m * LDT + half * 8);
      u.q[1] = *reinterpret_cast<const uint4*>(Ab + m * LDT + 16 + half * 8);
      af[i] = u.v;
    }
#pragma unroll
    for (int j = 0; j < 4; ++j) {
      const int n = waveN + j * 16 + l16;
      Q u;
      u.q[0] = *reinterpret_cast<const uint4*>(Bb + n * LDT + half * 8);
      u.q[1] = *reinterpret_cast<const uint4*>(Bb + n * LDT + 16 + half * 8);
      bfr[j] = u.v;
    }

    if (kt + 1 < ksteps) fill(1 - cur, kt + 1);  // overlap next fill with WMMA

#pragma unroll
    for (int i = 0; i < 4; ++i)
#pragma unroll
      for (int j = 0; j < 4; ++j)
        acc[i][j] = __builtin_amdgcn_wmma_f32_16x16x32_bf16(
            false, af[i], false, bfr[j], (short)0, acc[i][j], false, false);
  }

  // Epilogue. C layout: element (m,n): m = vgpr + 8*half, n = l16.
  const int gm0 = blockIdx.y * 128 + waveM;
  const int gn0 = blockIdx.x * 256 + waveN + l16;
#pragma unroll
  for (int i = 0; i < 4; ++i) {
#pragma unroll
    for (int j = 0; j < 4; ++j) {
      const int col = gn0 + j * 16;
#pragma unroll
      for (int v = 0; v < 8; ++v) {
        const int row = gm0 + i * 16 + half * 8 + v;
        float val = acc[i][j][v];
        if (GELU) {
          val = 0.5f * val * (1.0f + erff(val * 0.70710678118654752f));
          Cb[(size_t)row * N + col] = __float2bfloat16(val);
        } else {
          Cf[(size_t)row * N + col] = val;
        }
      }
    }
  }
}

// ---------------------------------------------------------------------------
extern "C" void kernel_launch(void* const* d_in, const int* in_sizes, int n_in,
                              void* d_out, int out_size, void* d_ws, size_t ws_size,
                              hipStream_t stream) {
  const float* x     = (const float*)d_in[0];   // [TOK, EMB]
  const float* sign  = (const float*)d_in[1];   // [NN]
  const int*   pidx  = (const int*)d_in[2];     // [2, NN]; row 0 = target rows
  const float* pvals = (const float*)d_in[3];   // [NN]
  const float* w_up  = (const float*)d_in[4];   // [HID, NN]
  const float* w_dn  = (const float*)d_in[5];   // [NN, HID]
  float* out = (float*)d_out;                   // [TOK, EMB]

  // Workspace layout (bytes):
  //   [0, 128MB)      w2T (f32 [HID,NN])  -- reused as hB (bf16 [TOK,HID]) after prep
  //   [128MB, 160MB)  xB  (bf16 [TOK,EMB])
  //   [160MB, 168MB)  upT (bf16 [HID,EMB])
  //   [168MB, 176MB)  dnB (bf16 [EMB,HID])
  char* ws = (char*)d_ws;
  float* w2T = (float*)(ws);
  bf16*  hB  = (bf16*)(ws);
  bf16*  xB  = (bf16*)(ws + (size_t)134217728);
  bf16*  upT = (bf16*)(ws + (size_t)134217728 + 33554432);
  bf16*  dnB = (bf16*)(ws + (size_t)134217728 + 33554432 + 8388608);

  const int* prows = pidx;  // proj_indices[0]

  // 1) transpose w_down so the down-path FWHT reads coalesced rows
  k_transpose<<<dim3(HID / 32, NN / 32), dim3(32, 8), 0, stream>>>(w_dn, w2T);
  // 2) fused FWHT + sparse scatter -> up^T (bf16, GEMM1 B operand)
  k_fwht_scatter<0><<<HID, 256, 0, stream>>>(w_up, sign, prows, pvals, upT);
  // 3) fused FWHT + sparse scatter -> down (bf16, GEMM2 B operand)
  k_fwht_scatter<1><<<HID, 256, 0, stream>>>(w2T, sign, prows, pvals, dnB);
  // 4) x -> bf16
  k_f32_to_bf16<<<(TOK * EMB / 4 + 255) / 256, 256, 0, stream>>>(
      (const float4*)x, (uint2*)xB, TOK * EMB / 4);
  // 5) h = gelu(x @ up)  (bf16 out, overwrites w2T region)
  k_gemm<true><<<dim3(HID / 256, TOK / 128), 256, 0, stream>>>(
      xB, upT, TOK, HID, EMB, nullptr, hB);
  // 6) out = h @ down^T  (f32 out)
  k_gemm<false><<<dim3(EMB / 256, TOK / 128), 256, 0, stream>>>(
      hB, dnB, TOK, EMB, HID, out, nullptr);
}
